// GraphSAGE_43671227466151
// MI455X (gfx1250) — compile-verified, matrix operand order checked by reading, not run on hardware
//
#include <hip/hip_runtime.h>

#define NNODES 50000
#define NEDGES 800000
#define FEAT   128

typedef __bf16 bf16_t;
typedef __attribute__((ext_vector_type(16))) __bf16 v16bf;
typedef __attribute__((ext_vector_type(8)))  __bf16 v8bf;
typedef __attribute__((ext_vector_type(8)))  float  v8f;

// f32 -> bf16 round-to-nearest-even via bit math (used in prep kernels and the GEMM epilogue)
static __device__ __forceinline__ bf16_t f2bf(float f) {
    union { float f; unsigned u; } x; x.f = f;
    unsigned r = x.u + 0x7FFFu + ((x.u >> 16) & 1u);
    union { unsigned short s; bf16_t b; } y; y.s = (unsigned short)(r >> 16);
    return y.b;
}
// bf16 -> f32 (exact): shift into high half
static __device__ __forceinline__ float bf2f(bf16_t b) {
    union { bf16_t b; unsigned short s; } x; x.b = b;
    union { unsigned u; float f; } y; y.u = ((unsigned)x.s) << 16;
    return y.f;
}

// ---------------- utility / prep kernels ----------------

__global__ void zero_kernel(float* __restrict__ p, long long n) {
    long long i = (long long)blockIdx.x * blockDim.x + threadIdx.x;
    long long stride = (long long)gridDim.x * blockDim.x;
    for (; i < n; i += stride) p[i] = 0.0f;
}

__global__ void deg_kernel(const long long* __restrict__ dst,
                           float* __restrict__ deg, int nE) {
    int e = blockIdx.x * blockDim.x + threadIdx.x;
    if (e >= nE) return;
    __hip_atomic_fetch_add(deg + dst[e], 1.0f, __ATOMIC_RELAXED, __HIP_MEMORY_SCOPE_AGENT);
}

__global__ void invdeg_kernel(float* __restrict__ deg, int n) {
    int i = blockIdx.x * blockDim.x + threadIdx.x;
    if (i >= n) return;
    deg[i] = 1.0f / fmaxf(deg[i], 1.0f);
}

// f32 -> bf16 bulk convert (features)
__global__ void cvt_bf16_kernel(const float* __restrict__ in,
                                bf16_t* __restrict__ out, long long n) {
    long long i = (long long)blockIdx.x * blockDim.x + threadIdx.x;
    long long stride = (long long)gridDim.x * blockDim.x;
    for (; i < n; i += stride) out[i] = f2bf(in[i]);
}

// hnbf[node][f] = bf16(acc[node][f] * invdeg[node])
__global__ void finalize_neigh_kernel(const float* __restrict__ acc,
                                      const float* __restrict__ invdeg,
                                      bf16_t* __restrict__ hnbf, long long n) {
    long long i = (long long)blockIdx.x * blockDim.x + threadIdx.x;
    long long stride = (long long)gridDim.x * blockDim.x;
    for (; i < n; i += stride) hnbf[i] = f2bf(acc[i] * invdeg[i >> 7]);
}

// Transpose W[128][dout] (f32, row-major) -> Wt[dout][128] (bf16)
__global__ void wtrans_kernel(const float* __restrict__ W,
                              bf16_t* __restrict__ Wt, int dout) {
    int i = blockIdx.x * blockDim.x + threadIdx.x;      // i = n*128 + k
    if (i >= dout * FEAT) return;
    int n = i >> 7, k = i & 127;
    Wt[i] = f2bf(W[(size_t)k * dout + n]);
}

// Edge scatter: acc[dst] += f32(hbf[src]). One thread per (edge, 8-feat chunk).
__global__ void scatter_kernel(const bf16_t* __restrict__ hbf,
                               const long long* __restrict__ src,
                               const long long* __restrict__ dst,
                               float* __restrict__ acc, int nE) {
    long long tid = (long long)blockIdx.x * blockDim.x + threadIdx.x;
    long long e = tid >> 4;                 // 16 chunks of 8 bf16 per 128-feat row
    if (e >= nE) return;
    int f = ((int)tid & 15) * 8;
    long long s = src[e], d = dst[e];
    const v8bf v = *(const v8bf*)(hbf + s * FEAT + f);  // one b128 gather
    float* p = acc + d * FEAT + f;
    #pragma unroll
    for (int i = 0; i < 8; ++i)
        __hip_atomic_fetch_add(p + i, bf2f(v[i]), __ATOMIC_RELAXED, __HIP_MEMORY_SCOPE_AGENT);
}

// ---------------- fused SAGE layer GEMM (pure bf16 WMMA) ----------------
// out = act( hbf @ Ws + hnbf @ Wn + bias ); all fragment loads are b128.
__global__ __launch_bounds__(128) void sage_gemm_kernel(
        const bf16_t* __restrict__ hbf,   // [N,128]
        const bf16_t* __restrict__ hnbf,  // [N,128]
        const bf16_t* __restrict__ Wst,   // [dout,128] transposed
        const bf16_t* __restrict__ Wnt,   // [dout,128] transposed
        const float*  __restrict__ bias,  // [dout]
        bf16_t* __restrict__ out_bf,      // [N,dout] (intermediate layers)
        float*  __restrict__ out_f32,     // [N,dout] (final layer) or nullptr
        int dout, int act)
{
    const int lane  = threadIdx.x & 31;
    const int wave  = threadIdx.x >> 5;
    const int m0    = blockIdx.x * 16;                  // node tile
    const int n0    = (blockIdx.y * 4 + wave) * 16;     // output-col tile
    const int mr    = lane & 15;
    const int khalf = lane >> 4;                        // K-half select
    const int row   = m0 + mr;
    const int ncol  = n0 + mr;                          // B column = lane[3:0]

    const bf16_t* hrow  = hbf  + (size_t)row  * FEAT;
    const bf16_t* nrow  = hnbf + (size_t)row  * FEAT;
    const bf16_t* wsrow = Wst  + (size_t)ncol * FEAT;
    const bf16_t* wnrow = Wnt  + (size_t)ncol * FEAT;

    v8f c = {};
    #pragma unroll
    for (int k0 = 0; k0 < FEAT; k0 += 32) {
        const int off0 = k0 + khalf * 8;        // VGPRs 0..3: pairs k..k+7
        const int off1 = k0 + 16 + khalf * 8;   // VGPRs 4..7: pairs k+16..k+23
        const v8bf aSlo = *(const v8bf*)(hrow  + off0);
        const v8bf aShi = *(const v8bf*)(hrow  + off1);
        const v8bf aNlo = *(const v8bf*)(nrow  + off0);
        const v8bf aNhi = *(const v8bf*)(nrow  + off1);
        const v8bf bSlo = *(const v8bf*)(wsrow + off0);
        const v8bf bShi = *(const v8bf*)(wsrow + off1);
        const v8bf bNlo = *(const v8bf*)(wnrow + off0);
        const v8bf bNhi = *(const v8bf*)(wnrow + off1);
        const v16bf aS = __builtin_shufflevector(aSlo, aShi, 0,1,2,3,4,5,6,7,8,9,10,11,12,13,14,15);
        const v16bf aN = __builtin_shufflevector(aNlo, aNhi, 0,1,2,3,4,5,6,7,8,9,10,11,12,13,14,15);
        const v16bf bS = __builtin_shufflevector(bSlo, bShi, 0,1,2,3,4,5,6,7,8,9,10,11,12,13,14,15);
        const v16bf bN = __builtin_shufflevector(bNlo, bNhi, 0,1,2,3,4,5,6,7,8,9,10,11,12,13,14,15);
        c = __builtin_amdgcn_wmma_f32_16x16x32_bf16(false, aS, false, bS,
                                                    (short)0, c, false, false);
        c = __builtin_amdgcn_wmma_f32_16x16x32_bf16(false, aN, false, bN,
                                                    (short)0, c, false, false);
    }

    // Epilogue. D layout: VGPR r -> M = r + 8*khalf, N = lane[3:0]
    const float bv = bias[ncol];
    const int mbase = m0 + (khalf ? 8 : 0);
    if (out_f32) {
        #pragma unroll
        for (int r = 0; r < 8; ++r) {
            float v = c[r] + bv;
            if (act) v = fmaxf(v, 0.0f);
            out_f32[(size_t)(mbase + r) * dout + ncol] = v;
        }
    } else {
        #pragma unroll
        for (int r = 0; r < 8; ++r) {
            float v = c[r] + bv;
            if (act) v = fmaxf(v, 0.0f);
            out_bf[(size_t)(mbase + r) * dout + ncol] = f2bf(v);
        }
    }
}

// ---------------- driver ----------------

extern "C" void kernel_launch(void* const* d_in, const int* in_sizes, int n_in,
                              void* d_out, int out_size, void* d_ws, size_t ws_size,
                              hipStream_t stream) {
    const float*     feat = (const float*)d_in[0];
    const long long* src  = (const long long*)d_in[1];
    const long long* dst  = (const long long*)d_in[2];
    const float* W[3][2] = { { (const float*)d_in[3], (const float*)d_in[4] },
                             { (const float*)d_in[6], (const float*)d_in[7] },
                             { (const float*)d_in[9], (const float*)d_in[10] } };
    const float* B[3] = { (const float*)d_in[5], (const float*)d_in[8], (const float*)d_in[11] };
    const int DOUT[3] = { 128, 128, 64 };

    const size_t NF = (size_t)NNODES * FEAT;

    // Workspace layout
    float* deg = (float*)d_ws;                     // [N]            f32
    float* acc = deg + NNODES;                     // [N,128]        f32
    bf16_t* hbf0 = (bf16_t*)(acc + NF);            // [N,128]        bf16
    bf16_t* hbf1 = hbf0 + NF;
    bf16_t* hbf2 = hbf1 + NF;
    bf16_t* hnbf = hbf2 + NF;                      // neighbor mean  bf16
    bf16_t* Wt[3][2];
    bf16_t* wp = hnbf + NF;
    for (int l = 0; l < 3; ++l)
        for (int s = 0; s < 2; ++s) { Wt[l][s] = wp; wp += (size_t)DOUT[l] * FEAT; }

    // One-time prep (per call; deterministic)
    zero_kernel<<<1024, 256, 0, stream>>>(deg, NNODES);
    deg_kernel<<<(NEDGES + 255) / 256, 256, 0, stream>>>(dst, deg, NEDGES);
    invdeg_kernel<<<(NNODES + 255) / 256, 256, 0, stream>>>(deg, NNODES);
    cvt_bf16_kernel<<<4096, 256, 0, stream>>>(feat, hbf0, (long long)NF);
    for (int l = 0; l < 3; ++l)
        for (int s = 0; s < 2; ++s)
            wtrans_kernel<<<(DOUT[l] * FEAT + 255) / 256, 256, 0, stream>>>(W[l][s], Wt[l][s], DOUT[l]);

    bf16_t* hin[3]  = { hbf0, hbf1, hbf2 };
    bf16_t* hout[3] = { hbf1, hbf2, nullptr };

    const long long scatterThreads = (long long)NEDGES * 16;
    const unsigned scatterBlocks = (unsigned)((scatterThreads + 255) / 256);

    for (int l = 0; l < 3; ++l) {
        zero_kernel<<<4096, 256, 0, stream>>>(acc, (long long)NF);
        scatter_kernel<<<scatterBlocks, 256, 0, stream>>>(hin[l], src, dst, acc, NEDGES);
        finalize_neigh_kernel<<<4096, 256, 0, stream>>>(acc, deg, hnbf, (long long)NF);
        dim3 grid(NNODES / 16, DOUT[l] / 64);
        sage_gemm_kernel<<<grid, 128, 0, stream>>>(hin[l], hnbf,
                                                   Wt[l][0], Wt[l][1], B[l],
                                                   hout[l],
                                                   (l == 2) ? (float*)d_out : nullptr,
                                                   DOUT[l], (l < 2) ? 1 : 0);
    }
}
